// RotaryMultiheadAttention_8959301779634
// MI455X (gfx1250) — matrix operand order repaired and checked by reading, not compiled, not run on hardware
//
#include <hip/hip_runtime.h>
#include <hip/hip_bf16.h>

#define D_MODEL  1024
#define NHEAD    16
#define HEAD_DIM 64
#define SEQ      2048
#define BATCH    4
#define NTOK     (BATCH * SEQ)   // 8192

typedef __bf16 v16bf __attribute__((ext_vector_type(16)));
typedef float  v8f   __attribute__((ext_vector_type(8)));

static __device__ __forceinline__ unsigned short f2bf(float f) {
    unsigned u = __float_as_uint(f);
    u += 0x7FFFu + ((u >> 16) & 1u);           // round to nearest even
    return (unsigned short)(u >> 16);
}
static __device__ __forceinline__ float bf2f(unsigned short h) {
    return __uint_as_float(((unsigned)h) << 16);
}

static __device__ __forceinline__ v8f wmma_bf16(v16bf a, v16bf b, v8f c) {
    return __builtin_amdgcn_wmma_f32_16x16x32_bf16(
        /*neg_a=*/false, a, /*neg_b=*/false, b,
        /*c_mod=*/(short)0, c, /*reuse_a=*/false, /*reuse_b=*/false);
}

// A-fragment (16x32 bf16): per lane, 16 contiguous bf16 at p.
static __device__ __forceinline__ v16bf ld_frag_split(const unsigned short* p) {
    v16bf r;
    ((uint4*)&r)[0] = *(const uint4*)(p);
    ((uint4*)&r)[1] = *(const uint4*)(p + 8);
    return r;
}
// B-fragment (32x16 bf16): per lane, 16 contiguous bf16 (K-major column).
static __device__ __forceinline__ v16bf ld_frag_contig(const unsigned short* p) {
    v16bf r;
    ((uint4*)&r)[0] = *(const uint4*)(p);
    ((uint4*)&r)[1] = *(const uint4*)(p + 8);
    return r;
}

// ---------------------------------------------------------------- convert (x4 vectorized)
__global__ __launch_bounds__(256) void cvt_f32_bf16_kernel(
        const float* __restrict__ src, unsigned short* __restrict__ dst, int n) {
    int i = (blockIdx.x * blockDim.x + threadIdx.x) * 4;
    if (i + 3 < n) {
        float4 f = *(const float4*)(src + i);
        ushort4 o;
        o.x = f2bf(f.x); o.y = f2bf(f.y); o.z = f2bf(f.z); o.w = f2bf(f.w);
        *(ushort4*)(dst + i) = o;
    }
}

// ---------------------------------------------------------------- RoPE (+ Q scale)
// Layout (b,h,s,d) bf16, in place. One thread per (b,h,s,j<32).
__global__ __launch_bounds__(256) void rope_kernel(
        unsigned short* __restrict__ Qh, unsigned short* __restrict__ Kh) {
    unsigned t = blockIdx.x * blockDim.x + threadIdx.x;
    int j  = t & 31;
    unsigned rest = t >> 5;
    int s  = rest % SEQ;
    int bh = rest / SEQ;                       // 0..63
    if (bh >= BATCH * NHEAD) return;
    float inv_freq = __expf(-(float)j * (__logf(10000.0f) / 32.0f));
    float ang = (float)s * inv_freq;
    float c = __cosf(ang), sn = __sinf(ang);
    size_t base = ((size_t)bh * SEQ + s) * HEAD_DIM;

    float q0 = bf2f(Qh[base + j]), q1 = bf2f(Qh[base + j + 32]);
    Qh[base + j]      = f2bf((q0 * c - q1 * sn) * 0.125f);  // fold 1/sqrt(64)
    Qh[base + j + 32] = f2bf((q1 * c + q0 * sn) * 0.125f);

    float k0 = bf2f(Kh[base + j]), k1 = bf2f(Kh[base + j + 32]);
    Kh[base + j]      = f2bf(k0 * c - k1 * sn);
    Kh[base + j + 32] = f2bf(k1 * c + k0 * sn);
}

// ---------------------------------------------------------------- GEMM: C = A @ W^T + bias
// A: (NTOK x 1024) bf16 row-major. W: (1024 x 1024) bf16 row-major (W^T col n = W row n).
// One wave computes a 32M x 64N tile. Explicit even/odd two-stage software pipeline:
// stage X's fragment loads are issued before stage Y's 8 WMMAs, no rotation copies.
// mode 0: bf16 out (b,h,s,d); mode 1: bf16 out (b,h,d,s); mode 2: fp32 out row-major.
__global__ __launch_bounds__(256) void gemm_bf16_kernel(
        const unsigned short* __restrict__ A,
        const unsigned short* __restrict__ W,
        const float* __restrict__ bias,
        unsigned short* __restrict__ out_bf,
        float* __restrict__ out_f, int mode) {
    const int lane = threadIdx.x & 31;
    const int mrow = lane & 15;
    const int hi   = lane >> 4;                        // lane-half
    const int wgid = blockIdx.x * (blockDim.x >> 5) + (threadIdx.x >> 5);
    const int NT_N = D_MODEL / 64;                     // 16
    const int mt = wgid / NT_N;
    const int nt = wgid % NT_N;
    if (mt >= NTOK / 32) return;

    const unsigned short* Ar[2];
    Ar[0] = A + (size_t)(mt * 32 + mrow) * D_MODEL + hi * 16;
    Ar[1] = A + (size_t)(mt * 32 + 16 + mrow) * D_MODEL + hi * 16;
    const unsigned short* Wr[4];
#pragma unroll
    for (int j = 0; j < 4; ++j)
        Wr[j] = W + (size_t)(nt * 64 + j * 16 + mrow) * D_MODEL + hi * 16;

    v8f acc[8] = {};
    v16bf a0[2], b0[4], a1[2], b1[4];

    // preload stage-0 fragments @ k = 0
#pragma unroll
    for (int mi = 0; mi < 2; ++mi) a0[mi] = ld_frag_split(Ar[mi]);
#pragma unroll
    for (int j = 0; j < 4; ++j)   b0[j]  = ld_frag_contig(Wr[j]);

    for (int k0 = 0; k0 < D_MODEL - 64; k0 += 64) {
        // stage-1 loads @ k0+32 (in flight under stage-0 WMMAs)
#pragma unroll
        for (int mi = 0; mi < 2; ++mi) a1[mi] = ld_frag_split(Ar[mi] + k0 + 32);
#pragma unroll
        for (int j = 0; j < 4; ++j)   b1[j]  = ld_frag_contig(Wr[j] + k0 + 32);
#pragma unroll
        for (int j = 0; j < 4; ++j)
#pragma unroll
            for (int mi = 0; mi < 2; ++mi)
                acc[mi * 4 + j] = wmma_bf16(a0[mi], b0[j], acc[mi * 4 + j]);
        // stage-0 loads @ k0+64 (in flight under stage-1 WMMAs)
#pragma unroll
        for (int mi = 0; mi < 2; ++mi) a0[mi] = ld_frag_split(Ar[mi] + k0 + 64);
#pragma unroll
        for (int j = 0; j < 4; ++j)   b0[j]  = ld_frag_contig(Wr[j] + k0 + 64);
#pragma unroll
        for (int j = 0; j < 4; ++j)
#pragma unroll
            for (int mi = 0; mi < 2; ++mi)
                acc[mi * 4 + j] = wmma_bf16(a1[mi], b1[j], acc[mi * 4 + j]);
    }
    // tail: stage-0 holds k = D_MODEL-64; load final stage-1 @ D_MODEL-32
#pragma unroll
    for (int mi = 0; mi < 2; ++mi) a1[mi] = ld_frag_split(Ar[mi] + D_MODEL - 32);
#pragma unroll
    for (int j = 0; j < 4; ++j)   b1[j]  = ld_frag_contig(Wr[j] + D_MODEL - 32);
#pragma unroll
    for (int j = 0; j < 4; ++j)
#pragma unroll
        for (int mi = 0; mi < 2; ++mi)
            acc[mi * 4 + j] = wmma_bf16(a0[mi], b0[j], acc[mi * 4 + j]);
#pragma unroll
    for (int j = 0; j < 4; ++j)
#pragma unroll
        for (int mi = 0; mi < 2; ++mi)
            acc[mi * 4 + j] = wmma_bf16(a1[mi], b1[j], acc[mi * 4 + j]);

    // epilogue
#pragma unroll
    for (int mi = 0; mi < 2; ++mi)
#pragma unroll
        for (int j = 0; j < 4; ++j) {
            int ncol = nt * 64 + j * 16 + mrow;
            float bv = bias[ncol];
#pragma unroll
            for (int r = 0; r < 8; ++r) {
                int row = mt * 32 + mi * 16 + r + hi * 8;     // token index
                float val = acc[mi * 4 + j][r] + bv;
                if (mode == 2) {
                    out_f[(size_t)row * D_MODEL + ncol] = val;
                } else {
                    int b = row / SEQ, s = row % SEQ;
                    int h = ncol / HEAD_DIM, d = ncol % HEAD_DIM;
                    size_t idx;
                    if (mode == 0)       // (b,h,s,d)
                        idx = (((size_t)(b * NHEAD + h)) * SEQ + s) * HEAD_DIM + d;
                    else                 // (b,h,d,s)  (V transposed)
                        idx = (((size_t)(b * NHEAD + h)) * HEAD_DIM + d) * SEQ + s;
                    out_bf[idx] = f2bf(val);
                }
            }
        }
}

// ---------------------------------------------------------------- flash attention
// Grid: (SEQ/64, NHEAD, BATCH), block 128 (4 waves). Wave w owns q rows [blk*64+w*16, +16).
// Waves are fully independent: the P-transpose LDS tile is wave-private, ordered with
// wave-local s_wait_dscnt (LDS ops are in-order within a wave), no block barrier needed.
__global__ __launch_bounds__(128) void attn_kernel(
        const unsigned short* __restrict__ Qh,
        const unsigned short* __restrict__ Kh,
        const unsigned short* __restrict__ Vt,
        unsigned short* __restrict__ Obuf) {
    __shared__ __align__(16) unsigned short Plds[4][16 * 32];

    const int lane = threadIdx.x & 31;
    const int wave = threadIdx.x >> 5;
    const int mrow = lane & 15;
    const int hi   = lane >> 4;
    const int b = blockIdx.z, h = blockIdx.y;
    const int q0 = blockIdx.x * 64 + wave * 16;

    const unsigned short* Qb = Qh + ((size_t)(b * NHEAD + h) * SEQ) * HEAD_DIM;
    const unsigned short* Kb = Kh + ((size_t)(b * NHEAD + h) * SEQ) * HEAD_DIM;
    const unsigned short* Vb = Vt + ((size_t)(b * NHEAD + h) * HEAD_DIM) * SEQ;

    // Q A-fragments for this wave's 16 rows (held in registers for the whole pass)
    v16bf qa[2];
#pragma unroll
    for (int ks = 0; ks < 2; ++ks)
        qa[ks] = ld_frag_split(Qb + (size_t)(q0 + mrow) * HEAD_DIM + ks * 32 + hi * 16);

    v8f o[4] = {};
    float mstat[8], lstat[8];
#pragma unroll
    for (int r = 0; r < 8; ++r) { mstat[r] = -1e30f; lstat[r] = 0.0f; }

    for (int kc = 0; kc < SEQ; kc += 32) {
        // ---- prefetch next chunk's K / V cachelines (global_prefetch_b8)
        if (kc + 32 < SEQ) {
            __builtin_prefetch(Kb + (size_t)(kc + 32 + mrow) * HEAD_DIM, 0, 3);
            __builtin_prefetch(Kb + (size_t)(kc + 48 + mrow) * HEAD_DIM, 0, 3);
#pragma unroll
            for (int dt = 0; dt < 4; ++dt)
                __builtin_prefetch(Vb + (size_t)(dt * 16 + mrow) * SEQ + kc + 32, 0, 3);
        }

        // ---- issue ALL fragment loads first: K frags feed the score WMMAs now,
        //      V frags stay in flight and are hidden under the softmax math.
        const unsigned short* k0p = Kb + (size_t)(kc + mrow) * HEAD_DIM + hi * 16;
        const unsigned short* k1p = Kb + (size_t)(kc + 16 + mrow) * HEAD_DIM + hi * 16;
        v16bf kb0a = ld_frag_contig(k0p);
        v16bf kb0b = ld_frag_contig(k0p + 32);
        v16bf kb1a = ld_frag_contig(k1p);
        v16bf kb1b = ld_frag_contig(k1p + 32);
        v16bf vb[4];
#pragma unroll
        for (int dt = 0; dt < 4; ++dt)
            vb[dt] = ld_frag_contig(Vb + (size_t)(dt * 16 + mrow) * SEQ + kc + hi * 16);

        // ---- scores: two 16x16 key tiles, contraction over head_dim (2 x 32)
        v8f s0 = {}, s1 = {};
        s0 = wmma_bf16(qa[0], kb0a, s0);
        s0 = wmma_bf16(qa[1], kb0b, s0);
        s1 = wmma_bf16(qa[0], kb1a, s1);
        s1 = wmma_bf16(qa[1], kb1b, s1);

        // ---- online softmax (rows r + hi*8 per lane, 16 key-columns across lanes)
        float scale[8], p0v[8], p1v[8];
#pragma unroll
        for (int r = 0; r < 8; ++r) {
            float rmax = fmaxf(s0[r], s1[r]);
#pragma unroll
            for (int off = 1; off < 16; off <<= 1)
                rmax = fmaxf(rmax, __shfl_xor(rmax, off, 32));
            float mn = fmaxf(mstat[r], rmax);
            scale[r] = __expf(mstat[r] - mn);
            mstat[r] = mn;
            p0v[r] = __expf(s0[r] - mn);
            p1v[r] = __expf(s1[r] - mn);
            float rs = p0v[r] + p1v[r];
#pragma unroll
            for (int off = 1; off < 16; off <<= 1)
                rs += __shfl_xor(rs, off, 32);
            lstat[r] = lstat[r] * scale[r] + rs;
        }
#pragma unroll
        for (int dt = 0; dt < 4; ++dt)
#pragma unroll
            for (int r = 0; r < 8; ++r)
                o[dt][r] *= scale[r];

        // ---- transpose P through the wave-private LDS tile (16 x 32 bf16, row-major)
#pragma unroll
        for (int r = 0; r < 8; ++r) {
            Plds[wave][(r + hi * 8) * 32 + mrow]      = f2bf(p0v[r]);
            Plds[wave][(r + hi * 8) * 32 + 16 + mrow] = f2bf(p1v[r]);
        }
        // wave-local ordering: LDS ops are in-order per wave; wait for stores, block
        // compiler reordering. No block-wide barrier needed (region is wave-private).
        asm volatile("s_wait_dscnt 0x0" ::: "memory");

        v16bf pa = ld_frag_split(&Plds[wave][mrow * 32 + hi * 16]);

        // ---- O += P @ V  (V stored (d, s): column d is key-contiguous)
#pragma unroll
        for (int dt = 0; dt < 4; ++dt)
            o[dt] = wmma_bf16(pa, vb[dt], o[dt]);
        asm volatile("" ::: "memory");
    }

    // ---- normalize and store O as (b, s, h*64+d) bf16
#pragma unroll
    for (int r = 0; r < 8; ++r) {
        float inv = 1.0f / lstat[r];
        size_t row = (size_t)b * SEQ + q0 + r + hi * 8;
#pragma unroll
        for (int dt = 0; dt < 4; ++dt)
            Obuf[row * D_MODEL + h * HEAD_DIM + dt * 16 + mrow] = f2bf(o[dt][r] * inv);
    }
}

// ---------------------------------------------------------------- launch
extern "C" void kernel_launch(void* const* d_in, const int* in_sizes, int n_in,
                              void* d_out, int out_size, void* d_ws, size_t ws_size,
                              hipStream_t stream) {
    const float* q  = (const float*)d_in[0];
    const float* k  = (const float*)d_in[1];
    const float* v  = (const float*)d_in[2];
    const float* Wq = (const float*)d_in[3];
    const float* bq = (const float*)d_in[4];
    const float* Wk = (const float*)d_in[5];
    const float* bk = (const float*)d_in[6];
    const float* Wv = (const float*)d_in[7];
    const float* bv = (const float*)d_in[8];
    const float* Wo = (const float*)d_in[9];
    const float* bo = (const float*)d_in[10];

    const size_t NX = (size_t)NTOK * D_MODEL;      // 8388608 elems
    const size_t NW = (size_t)D_MODEL * D_MODEL;   // 1048576 elems

    unsigned short* Xq   = (unsigned short*)d_ws;
    unsigned short* Xk   = Xq + NX;
    unsigned short* Xv   = Xk + NX;
    unsigned short* Wqb  = Xv + NX;
    unsigned short* Wkb  = Wqb + NW;
    unsigned short* Wvb  = Wkb + NW;
    unsigned short* Wob  = Wvb + NW;
    unsigned short* Qh   = Wob + NW;
    unsigned short* Kh   = Qh + NX;
    unsigned short* Vt   = Kh + NX;
    unsigned short* Obuf = Vt + NX;

    // 1) fp32 -> bf16 (4 elems/thread)
    cvt_f32_bf16_kernel<<<(int)(NX / 1024), 256, 0, stream>>>(q, Xq, (int)NX);
    cvt_f32_bf16_kernel<<<(int)(NX / 1024), 256, 0, stream>>>(k, Xk, (int)NX);
    cvt_f32_bf16_kernel<<<(int)(NX / 1024), 256, 0, stream>>>(v, Xv, (int)NX);
    cvt_f32_bf16_kernel<<<(int)(NW / 1024), 256, 0, stream>>>(Wq, Wqb, (int)NW);
    cvt_f32_bf16_kernel<<<(int)(NW / 1024), 256, 0, stream>>>(Wk, Wkb, (int)NW);
    cvt_f32_bf16_kernel<<<(int)(NW / 1024), 256, 0, stream>>>(Wv, Wvb, (int)NW);
    cvt_f32_bf16_kernel<<<(int)(NW / 1024), 256, 0, stream>>>(Wo, Wob, (int)NW);

    // 2) projections  (4096 wave-tiles of 32x64 -> 512 blocks of 8 waves)
    const int gemm_blocks = (NTOK / 32) * (D_MODEL / 64) / 8;
    gemm_bf16_kernel<<<gemm_blocks, 256, 0, stream>>>(Xq, Wqb, bq, Qh, nullptr, 0);
    gemm_bf16_kernel<<<gemm_blocks, 256, 0, stream>>>(Xk, Wkb, bk, Kh, nullptr, 0);
    gemm_bf16_kernel<<<gemm_blocks, 256, 0, stream>>>(Xv, Wvb, bv, Vt, nullptr, 1);

    // 3) RoPE on Q,K (and fold 1/sqrt(head_dim) into Q)
    const int rope_threads = BATCH * NHEAD * SEQ * 32;
    rope_kernel<<<rope_threads / 256, 256, 0, stream>>>(Qh, Kh);

    // 4) attention
    dim3 ag(SEQ / 64, NHEAD, BATCH);
    attn_kernel<<<ag, 128, 0, stream>>>(Qh, Kh, Vt, Obuf);

    // 5) output projection -> fp32 d_out
    gemm_bf16_kernel<<<gemm_blocks, 256, 0, stream>>>(Obuf, Wob, bo, nullptr,
                                                      (float*)d_out, 2);
}
